// SwinBlock_41266045780057
// MI455X (gfx1250) — matrix operand order repaired and checked by reading, not compiled, not run on hardware
//
#include <hip/hip_runtime.h>
#include <hip/hip_bf16.h>

typedef __attribute__((ext_vector_type(16))) _Float16 v16h;
typedef __attribute__((ext_vector_type(8)))  _Float16 v8h;
typedef __attribute__((ext_vector_type(8)))  float    v8f;
typedef __attribute__((ext_vector_type(4)))  int      v4i;

#define SHUF16(lo, hi) __builtin_shufflevector(lo, hi, 0,1,2,3,4,5,6,7,8,9,10,11,12,13,14,15)

// ---------------------------------------------------------------------------
// Async global->LDS (CDNA5 GLOBAL_LOAD_ASYNC_TO_LDS_B128, ASYNCcnt-tracked).
// Builtin signature (from hipcc diagnostic): (int4 AS1*, int4 AS3*, imm, imm).
// Guarded: falls back to VGPR-staged copies if the builtin is unavailable.
// LDS pointer arg: flat addr low 32 bits == LDS byte offset (ISA 10.2 aperture).
// ---------------------------------------------------------------------------
#if defined(__gfx1250__) && __has_builtin(__builtin_amdgcn_global_load_async_to_lds_b128) && \
    __has_builtin(__builtin_amdgcn_s_wait_asynccnt)
#define USE_ASYNC_LDS 1
#else
#define USE_ASYNC_LDS 0
#endif

static __device__ __forceinline__ void copy16_g2l(const _Float16* g, _Float16* l) {
#if USE_ASYNC_LDS
  typedef __attribute__((address_space(1))) v4i* gp_t;
  typedef __attribute__((address_space(3))) v4i* lp_t;
  __builtin_amdgcn_global_load_async_to_lds_b128(
      (gp_t)(unsigned long long)g,
      (lp_t)(unsigned int)(unsigned long long)l, 0, 0);
#else
  *(v8h*)l = *(const v8h*)g;
#endif
}
static __device__ __forceinline__ void stage_fence() {
#if USE_ASYNC_LDS
  __builtin_amdgcn_s_wait_asynccnt(0);
#endif
  __syncthreads();
}

// ---------------------------------------------------------------------------
// Fragment loaders (per CDNA5 ISA 7.12.2 VGPR layouts, wave32)
// A (16x32 f16): lane m=l&15; halves 0..7 = K[koff..+7], 8..15 = K[koff+16..+23], koff=8*(l>>4)
// B (32x16 f16) from K-major (Bt[n][k]) tile: lane n=l&15; halves 0..7 = K[koff..+7], 8..15 = K[koff+16..+23], koff=8*(l>>4)
// C/D (16x16 f32): elem r of v8f -> M = r + 8*(l>>4), N = l&15
// ---------------------------------------------------------------------------
static __device__ __forceinline__ v16h lds_load_a_frag(const _Float16* base, int ld) {
  const int lane = threadIdx.x & 31;
  const _Float16* p = base + (lane & 15) * ld + ((lane >> 4) << 3);
  v8h lo = *(const v8h*)p;
  v8h hi = *(const v8h*)(p + 16);
  return SHUF16(lo, hi);
}
static __device__ __forceinline__ v16h lds_load_b_frag(const _Float16* base, int ld) {
  const int lane = threadIdx.x & 31;
  const _Float16* p = base + (lane & 15) * ld + ((lane >> 4) << 3);
  v8h lo = *(const v8h*)p;
  v8h hi = *(const v8h*)(p + 16);
  return SHUF16(lo, hi);
}
static __device__ __forceinline__ v8f wmma_f16(v16h a, v16h b, v8f c) {
  return __builtin_amdgcn_wmma_f32_16x16x32_f16(false, a, false, b, (short)0, c, false, false);
}

// window mapping: token m (0..262143), channel c -> flat index into (4,384,256,256)
static __device__ __forceinline__ size_t xmap(int m, int c) {
  int win = m >> 6, tok = m & 63;
  int b = win >> 10, rem = win & 1023;
  int wy = rem >> 5, wx = rem & 31;
  int hy = (wy << 3) + (tok >> 3);
  int hx = (wx << 3) + (tok & 7);
  return ((size_t)(b * 384 + c) << 16) + ((size_t)hy << 8) + (size_t)hx;
}

// ---------------------------------------------------------------------------
// Weight pre-convert: W_f32[K,N] -> Wt_f16[N,K] (K-major == WMMA Bt layout)
// ---------------------------------------------------------------------------
__global__ __launch_bounds__(256) void wcvt_kernel(
    const float* __restrict__ w, _Float16* __restrict__ wt, int K, int N) {
  int idx = blockIdx.x * 256 + threadIdx.x;
  if (idx < K * N) {
    int k = idx / N, n = idx - k * N;            // coalesced read over n
    wt[(size_t)n * K + k] = (_Float16)w[idx];
  }
}

// ---------------------------------------------------------------------------
// LN1: token-per-lane, two-pass (dense 128B-line reads of channel planes)
// ---------------------------------------------------------------------------
__global__ __launch_bounds__(256) void ln1_kernel(
    const float* __restrict__ x, const float* __restrict__ g,
    const float* __restrict__ b, _Float16* __restrict__ h) {
  const int m = blockIdx.x * 256 + threadIdx.x;
  const size_t base = xmap(m, 0);
  float s = 0.f, ss = 0.f;
  for (int c = 0; c < 384; ++c) {
    float v = x[base + ((size_t)c << 16)];
    s += v; ss += v * v;
  }
  float mu = s * (1.f / 384.f);
  float var = ss * (1.f / 384.f) - mu * mu;
  float rstd = rsqrtf(var + 1e-5f);
  _Float16* hp = h + (size_t)m * 384;
  for (int c = 0; c < 384; ++c) {
    float v = x[base + ((size_t)c << 16)];
    hp[c] = (_Float16)((v - mu) * rstd * g[c] + b[c]);
  }
}

// LN2: wave per token over row-major t (coalesced both ways)
__global__ __launch_bounds__(256) void ln2_kernel(
    const float* __restrict__ t, const float* __restrict__ g,
    const float* __restrict__ b, _Float16* __restrict__ h) {
  const int wave = threadIdx.x >> 5, lane = threadIdx.x & 31;
  const int m = blockIdx.x * 8 + wave;
  float vals[12];
  float s = 0.f;
  #pragma unroll
  for (int j = 0; j < 12; ++j) {
    int c = lane + 32 * j;
    vals[j] = t[(size_t)m * 384 + c];
    s += vals[j];
  }
  #pragma unroll
  for (int off = 16; off; off >>= 1) s += __shfl_xor(s, off, 32);
  float mu = s * (1.f / 384.f);
  float vv = 0.f;
  #pragma unroll
  for (int j = 0; j < 12; ++j) { float d = vals[j] - mu; vv += d * d; }
  #pragma unroll
  for (int off = 16; off; off >>= 1) vv += __shfl_xor(vv, off, 32);
  float rstd = rsqrtf(vv * (1.f / 384.f) + 1e-5f);
  #pragma unroll
  for (int j = 0; j < 12; ++j) {
    int c = lane + 32 * j;
    h[(size_t)m * 384 + c] = (_Float16)((vals[j] - mu) * rstd * g[c] + b[c]);
  }
}

// ---------------------------------------------------------------------------
// WMMA GEMM: D[M,N] = A_f16[M,K] @ Wt_f16[N,K]^T + bias, custom epilogue.
// Block = 256 thr (8 waves), tile 256x64, k-step 32, double-buffered LDS with
// async global->LDS staging overlapping the WMMA chain.
// Wave w owns rows 32w..32w+31 (two 16-row strips) -> 8 WMMAs per k-step.
// ---------------------------------------------------------------------------
template<int K, class Epi>
__global__ __launch_bounds__(256) void gemm_kernel(
    const _Float16* __restrict__ A, const _Float16* __restrict__ Bt,
    const float* __restrict__ bias, int N, Epi epi) {
  __shared__ __align__(16) _Float16 As[2][256][40];
  __shared__ __align__(16) _Float16 Bs[2][64][40];   // K-major: Bs[buf][n][k]
  const int tid = threadIdx.x;
  const int wave = tid >> 5, lane = tid & 31;
  const int mBase = blockIdx.x * 256;
  const int nBase = blockIdx.y * 64;

  auto stage = [&](int sb, int kb) {
    #pragma unroll
    for (int i = 0; i < 4; ++i) {                 // A: 256x32 halfs, 16B chunks
      int idx = tid + i * 256;                    // 0..1023
      int r = idx >> 2, cc = (idx & 3) << 3;
      copy16_g2l(A + (size_t)(mBase + r) * K + kb + cc, &As[sb][r][cc]);
    }
    {                                             // B: 64x32 halfs, 16B chunks
      int n = tid >> 2, cc = (tid & 3) << 3;
      copy16_g2l(Bt + (size_t)(nBase + n) * K + kb + cc, &Bs[sb][n][cc]);
    }
  };

  v8f acc[2][4] = {};
  stage(0, 0);
  stage_fence();
  int buf = 0;
  for (int kb = 0; kb < K; kb += 32, buf ^= 1) {
    if (kb + 32 < K) stage(buf ^ 1, kb + 32);     // async-prefetch next tile
    v16h a0 = lds_load_a_frag(&As[buf][wave * 32][0], 40);
    v16h a1 = lds_load_a_frag(&As[buf][wave * 32 + 16][0], 40);
    #pragma unroll
    for (int nt = 0; nt < 4; ++nt) {
      v16h bf = lds_load_b_frag(&Bs[buf][nt * 16][0], 40);
      acc[0][nt] = wmma_f16(a0, bf, acc[0][nt]);
      acc[1][nt] = wmma_f16(a1, bf, acc[1][nt]);
    }
    stage_fence();
  }
  const int n0 = nBase + (lane & 15);
  #pragma unroll
  for (int s = 0; s < 2; ++s) {
    const int m0 = mBase + wave * 32 + s * 16 + ((lane >> 4) << 3);
    #pragma unroll
    for (int nt = 0; nt < 4; ++nt) {
      int n = n0 + nt * 16;
      float bv = bias[n];
      #pragma unroll
      for (int r = 0; r < 8; ++r) epi(m0 + r, n, acc[s][nt][r] + bv);
    }
  }
}

struct QkvEpi {                       // scale folded into Q columns
  _Float16* out; float scale;
  __device__ void operator()(int m, int n, float v) const {
    if (n < 384) v *= scale;
    out[(size_t)m * 1152 + n] = (_Float16)v;
  }
};
struct ProjEpi {                      // + windowed-x residual -> t (f32)
  float* t; const float* x;
  __device__ void operator()(int m, int n, float v) const {
    t[(size_t)m * 384 + n] = v + x[xmap(m, n)];
  }
};
struct Mlp1Epi {                      // exact GELU
  _Float16* out;
  __device__ void operator()(int m, int n, float v) const {
    out[(size_t)m * 1536 + n] = (_Float16)(0.5f * v * (1.f + erff(v * 0.70710678118654752f)));
  }
};
struct Mlp2Epi {                      // + t residual, window-reverse scatter
  float* y; const float* t;
  __device__ void operator()(int m, int n, float v) const {
    y[xmap(m, n)] = v + t[(size_t)m * 384 + n];
  }
};

// ---------------------------------------------------------------------------
// Attention: one block (4 waves) per (window, head). S=QK^T via WMMA (K tile in
// LDS is already Bt layout), bias+softmax in registers (rows wave-private),
// P f16 staged through LDS, O=PV via WMMA with V^T tile.
// ---------------------------------------------------------------------------
__global__ __launch_bounds__(128) void attn_kernel(
    const _Float16* __restrict__ qkv, const float* __restrict__ rpb_table,
    const int* __restrict__ rpb_index, _Float16* __restrict__ out) {
  const int win = blockIdx.x, head = blockIdx.y;
  const int tid = threadIdx.x, wave = tid >> 5, lane = tid & 31;
  __shared__ __align__(16) _Float16 Kl[64][40];   // Kl[tok][d] == Bt for Q@K^T
  __shared__ __align__(16) _Float16 Vt[32][72];   // Vt[d][tok] == Bt for P@V
  __shared__ __align__(16) _Float16 P[64][72];
  __shared__ float Bias[64][64];
  const size_t qbase = (size_t)win * 64 * 1152 + (size_t)head * 32;
  // K tile: straight 16B-chunk copies (already in Bt layout)
  for (int i = tid; i < 256; i += 128) {
    int tok = i >> 2, cc = (i & 3) << 3;
    *(v8h*)&Kl[tok][cc] = *(const v8h*)(qkv + qbase + (size_t)tok * 1152 + 384 + cc);
  }
  // V tile: 16B-chunk loads, transpose-scatter into Vt[d][tok]
  for (int i = tid; i < 256; i += 128) {
    int tok = i >> 2, cc = (i & 3) << 3;
    v8h v = *(const v8h*)(qkv + qbase + (size_t)tok * 1152 + 768 + cc);
    #pragma unroll
    for (int j = 0; j < 8; ++j) Vt[cc + j][tok] = v[j];
  }
  for (int i = tid; i < 4096; i += 128)
    Bias[i >> 6][i & 63] = rpb_table[rpb_index[i] * 12 + head];
  __syncthreads();

  const int l15 = lane & 15, hl = lane >> 4;
  const int mrow = wave * 16;
  v8f s[4] = {};
  {
    // Q fragment straight from global (already f16, pre-scaled)
    const _Float16* qp = qkv + qbase + (size_t)(mrow + l15) * 1152 + (hl << 3);
    v8h lo = *(const v8h*)qp;
    v8h hi = *(const v8h*)(qp + 16);
    v16h af = SHUF16(lo, hi);
    #pragma unroll
    for (int nt = 0; nt < 4; ++nt) {
      v16h bf = lds_load_b_frag(&Kl[nt * 16][0], 40);
      s[nt] = wmma_f16(af, bf, s[nt]);
    }
  }
  const int rbase = mrow + (hl << 3);
  #pragma unroll
  for (int r = 0; r < 8; ++r) {
    float e0 = s[0][r] + Bias[rbase + r][l15];
    float e1 = s[1][r] + Bias[rbase + r][16 + l15];
    float e2 = s[2][r] + Bias[rbase + r][32 + l15];
    float e3 = s[3][r] + Bias[rbase + r][48 + l15];
    float mx = fmaxf(fmaxf(e0, e1), fmaxf(e2, e3));
    #pragma unroll
    for (int off = 8; off; off >>= 1) mx = fmaxf(mx, __shfl_xor(mx, off, 32));
    e0 = __expf(e0 - mx); e1 = __expf(e1 - mx);
    e2 = __expf(e2 - mx); e3 = __expf(e3 - mx);
    float ss = e0 + e1 + e2 + e3;
    #pragma unroll
    for (int off = 8; off; off >>= 1) ss += __shfl_xor(ss, off, 32);
    float inv = 1.0f / ss;
    P[rbase + r][l15]      = (_Float16)(e0 * inv);
    P[rbase + r][16 + l15] = (_Float16)(e1 * inv);
    P[rbase + r][32 + l15] = (_Float16)(e2 * inv);
    P[rbase + r][48 + l15] = (_Float16)(e3 * inv);
  }
  // O = P @ V  (rows of P used here were written by this same wave)
  v8f o[2] = {};
  #pragma unroll
  for (int kb = 0; kb < 2; ++kb) {
    const _Float16* pp = &P[mrow + l15][kb * 32 + (hl << 3)];
    v8h lo = *(const v8h*)pp;
    v8h hi = *(const v8h*)(pp + 16);
    v16h af = SHUF16(lo, hi);
    #pragma unroll
    for (int nt = 0; nt < 2; ++nt) {
      v16h bf = lds_load_b_frag(&Vt[nt * 16][kb * 32], 72);
      o[nt] = wmma_f16(af, bf, o[nt]);
    }
  }
  #pragma unroll
  for (int nt = 0; nt < 2; ++nt) {
    int d = nt * 16 + l15;
    #pragma unroll
    for (int r = 0; r < 8; ++r) {
      int tok = rbase + r;
      out[(size_t)(win * 64 + tok) * 384 + head * 32 + d] = (_Float16)o[nt][r];
    }
  }
}

// ---------------------------------------------------------------------------
extern "C" void kernel_launch(void* const* d_in, const int* in_sizes, int n_in,
                              void* d_out, int out_size, void* d_ws, size_t ws_size,
                              hipStream_t stream) {
  (void)in_sizes; (void)n_in; (void)out_size; (void)ws_size;
  const float* x      = (const float*)d_in[0];
  const float* qkv_w  = (const float*)d_in[1];
  const float* qkv_b  = (const float*)d_in[2];
  const float* proj_w = (const float*)d_in[3];
  const float* proj_b = (const float*)d_in[4];
  const float* rpb_t  = (const float*)d_in[5];
  const int*   rpb_i  = (const int*)d_in[6];
  const float* ln1_g  = (const float*)d_in[7];
  const float* ln1_b  = (const float*)d_in[8];
  const float* ln2_g  = (const float*)d_in[9];
  const float* ln2_b  = (const float*)d_in[10];
  const float* w1     = (const float*)d_in[11];
  const float* b1     = (const float*)d_in[12];
  const float* w2     = (const float*)d_in[13];
  const float* b2     = (const float*)d_in[14];
  float* y = (float*)d_out;

  const size_t M = 262144;  // 4 * 32 * 32 * 64 tokens (windowed layout)
  _Float16* qkvbuf = (_Float16*)d_ws;             // M*1152 f16 (later reused as h2)
  _Float16* abuf   = qkvbuf + M * 1152;           // M*1536 f16 (h1 -> attn_out -> g)
  float*    tbuf   = (float*)(abuf + M * 1536);   // M*384 f32 residual stream
  _Float16* wq_t   = (_Float16*)(tbuf + M * 384); // f16 K-major weights
  _Float16* wp_t   = wq_t + (size_t)1152 * 384;
  _Float16* w1_t   = wp_t + (size_t)384 * 384;
  _Float16* w2_t   = w1_t + (size_t)1536 * 384;
  _Float16* h2buf  = qkvbuf;                      // qkv dead after attention

  const float scale = 0.17677669529663687f;       // 1/sqrt(32)

  wcvt_kernel<<<dim3((384 * 1152 + 255) / 256), 256, 0, stream>>>(qkv_w, wq_t, 384, 1152);
  wcvt_kernel<<<dim3((384 * 384 + 255) / 256), 256, 0, stream>>>(proj_w, wp_t, 384, 384);
  wcvt_kernel<<<dim3((384 * 1536 + 255) / 256), 256, 0, stream>>>(w1, w1_t, 384, 1536);
  wcvt_kernel<<<dim3((1536 * 384 + 255) / 256), 256, 0, stream>>>(w2, w2_t, 1536, 384);

  ln1_kernel<<<dim3(M / 256), 256, 0, stream>>>(x, ln1_g, ln1_b, abuf);
  gemm_kernel<384, QkvEpi><<<dim3(M / 256, 18), 256, 0, stream>>>(
      abuf, wq_t, qkv_b, 1152, QkvEpi{qkvbuf, scale});
  attn_kernel<<<dim3(4096, 12), 128, 0, stream>>>(qkvbuf, rpb_t, rpb_i, abuf);
  gemm_kernel<384, ProjEpi><<<dim3(M / 256, 6), 256, 0, stream>>>(
      abuf, wp_t, proj_b, 384, ProjEpi{tbuf, x});
  ln2_kernel<<<dim3(M / 8), 256, 0, stream>>>(tbuf, ln2_g, ln2_b, h2buf);
  gemm_kernel<384, Mlp1Epi><<<dim3(M / 256, 24), 256, 0, stream>>>(
      h2buf, w1_t, b1, 1536, Mlp1Epi{abuf});
  gemm_kernel<1536, Mlp2Epi><<<dim3(M / 256, 6), 256, 0, stream>>>(
      abuf, w2_t, b2, 384, Mlp2Epi{y, tbuf});
}
